// SimpleGCN_22935125360907
// MI455X (gfx1250) — matrix operand order repaired and checked by reading, not compiled, not run on hardware
//
#include <hip/hip_runtime.h>

// ---------- types ----------
typedef __attribute__((ext_vector_type(16))) __bf16 v16bf;
typedef __attribute__((ext_vector_type(8)))  float  v8f;

union Frag32B {                 // 32-byte fragment: 16 bf16 per lane
    v16bf v;
    uint4 q[2];
    unsigned short us[16];
};

// float -> bf16 (round to nearest even), raw bits
__device__ __forceinline__ unsigned short f2bf(float x) {
    unsigned u = __float_as_uint(x);
    u += 0x7FFFu + ((u >> 16) & 1u);
    return (unsigned short)(u >> 16);
}

// ---------- stage 1: column sums of (A + I) -> dinv ----------
__global__ void colsum_partial(const float* __restrict__ A, float* __restrict__ partial,
                               int n, int rowsPer) {
    int j = blockIdx.x * blockDim.x + threadIdx.x;
    int p = blockIdx.y;
    if (j >= n) return;
    int i0 = p * rowsPer;
    int i1 = min(n, i0 + rowsPer);
    float s = 0.f;
    for (int i = i0; i < i1; ++i) s += A[(size_t)i * n + j];   // coalesced over j
    partial[(size_t)p * n + j] = s;
}

__global__ void finish_dinv(const float* __restrict__ partial, float* __restrict__ dinv,
                            int n, int nparts) {
    int j = blockIdx.x * blockDim.x + threadIdx.x;
    if (j >= n) return;
    float s = 1.0f;                                            // self-loop
    for (int p = 0; p < nparts; ++p) s += partial[(size_t)p * n + j];
    dinv[j] = rsqrtf(s);                                       // s >= 1 always
}

// ---------- stage 2: Mt[j,i] = bf16( dinv_j * (A[i,j] + (i==j)) * dinv_i ) ----------
// Mt is P x P (P = padded N); pad rows/cols are written as exact zeros so the
// GEMM can run guard-free over the padded shape.
__global__ __launch_bounds__(256) void norm_transpose(const float* __restrict__ A,
                                                      const float* __restrict__ dinv,
                                                      unsigned short* __restrict__ Mt,
                                                      int n, int P) {
    __shared__ float tile[32][33];
    int i0 = blockIdx.x * 32, j0 = blockIdx.y * 32;
    int tx = threadIdx.x & 31, ty = threadIdx.x >> 5;          // 32 x 8
    #pragma unroll
    for (int r = 0; r < 32; r += 8) {
        int i = i0 + ty + r, j = j0 + tx;
        float v = 0.f;
        if (i < n && j < n) {
            v = A[(size_t)i * n + j];
            if (i == j) v += 1.0f;
        }
        tile[ty + r][tx] = v;
    }
    __syncthreads();
    #pragma unroll
    for (int r = 0; r < 32; r += 8) {
        int j = j0 + ty + r;   // output row
        int i = i0 + tx;       // output col
        float v = tile[tx][ty + r];
        unsigned short o = 0;
        if (i < n && j < n) o = f2bf(v * dinv[i] * dinv[j]);
        Mt[(size_t)j * P + i] = o;                             // always in padded range
    }
}

// ---------- stage 3: fp32 -> bf16 with zero row padding ----------
__global__ void cast_pad_bf16(const float* __restrict__ in, unsigned short* __restrict__ out,
                              int rows, int cols, int rowsPad) {
    size_t total = (size_t)rowsPad * cols;
    size_t i = (size_t)blockIdx.x * blockDim.x + threadIdx.x;
    size_t stride = (size_t)gridDim.x * blockDim.x;
    for (; i < total; i += stride) {
        int r = (int)(i / (unsigned)cols);
        out[i] = (r < rows) ? f2bf(in[i]) : (unsigned short)0;
    }
}

// ---------- WMMA bf16 GEMM:  C[Mpad,Nc] = A[Mpad,K] @ B[K,Nc] ----------
// Guard-free: A has Mpad rows (mult of 128, zero-padded), K mult of 32,
// B has K rows, Nc mult of 128. Block tile 128x128, BK=32, double-buffered LDS.
// 256 threads = 8 wave32s (4x2); each wave owns 32x64 = 2x4 WMMA tiles.
// EPI: 0 = plain -> bf16, 1 = bias+relu -> bf16, 2 = bias+relu -> fp32 (row<M guard).

__device__ __forceinline__ v16bf load_frag_a(const unsigned short* lds, int lane) {
    // A-matrix 16x32 bf16 per ISA table: lane m=lane&15, half=lane>>4,
    // elems 0-7: K = half*8..+7 ; elems 8-15: K = 16+half*8..+7
    int m = lane & 15, half = lane >> 4;
    Frag32B f;
    f.q[0] = *reinterpret_cast<const uint4*>(lds + m * 32 + half * 8);
    f.q[1] = *reinterpret_cast<const uint4*>(lds + m * 32 + 16 + half * 8);
    return f.v;
}

__device__ __forceinline__ v16bf load_frag_b(const unsigned short* ldsBt, int lane) {
    // B staged transposed as [n][k] (stride 32): lane n=lane&15 reads
    // elems 0-7: K = half*8..+7 ; elems 8-15: K = 16+half*8..+7
    int n = lane & 15, half = lane >> 4;
    Frag32B f;
    const uint4* p = reinterpret_cast<const uint4*>(ldsBt + n * 32 + half * 8);
    f.q[0] = p[0];
    f.q[1] = p[2];
    return f.v;
}

template <int EPI>
__global__ __launch_bounds__(256) void gemm_bf16_wmma(const unsigned short* __restrict__ A,
                                                      const unsigned short* __restrict__ B,
                                                      const float* __restrict__ bias,
                                                      unsigned short* __restrict__ Cb,
                                                      float* __restrict__ Cf,
                                                      int M, int K, int Nc) {
    __shared__ __align__(16) unsigned short sA[2][128 * 32];  // [m][k]        8 KB x2
    __shared__ __align__(16) unsigned short sB[2][128 * 32];  // [n][k] transp 8 KB x2

    const int tid  = threadIdx.x;
    const int lane = tid & 31;
    const int wave = tid >> 5;
    const int wm   = wave & 3;   // 0..3 -> 32-row stripe
    const int wn   = wave >> 2;  // 0..1 -> 64-col stripe
    const int bm   = blockIdx.x * 128;
    const int bn   = blockIdx.y * 128;

    // loader mapping (branch-free, all accesses in padded bounds)
    const int ar  = tid >> 1;             // A row   0..127
    const int ac  = (tid & 1) * 16;       // A col   0 or 16
    const int bkr = tid >> 3;             // B row   0..31
    const int bnc = (tid & 7) * 16;       // B col   0..112 step 16

    const unsigned short* gA = A + (size_t)(bm + ar) * K + ac;
    const unsigned short* gB = B + (size_t)bkr * Nc + bn + bnc;

    v8f acc[2][4] = {};

    auto loadTile = [&](int k0, int buf) {
        const uint4* a4 = reinterpret_cast<const uint4*>(gA + k0);
        uint4 av0 = a4[0], av1 = a4[1];
        if (k0 + 64 < K) __builtin_prefetch(gA + k0 + 64, 0, 1);  // global_prefetch_b8
        *reinterpret_cast<uint4*>(&sA[buf][ar * 32 + ac])     = av0;
        *reinterpret_cast<uint4*>(&sA[buf][ar * 32 + ac + 8]) = av1;
        Frag32B t;
        const uint4* b4 = reinterpret_cast<const uint4*>(gB + (size_t)k0 * Nc);
        t.q[0] = b4[0];
        t.q[1] = b4[1];
        #pragma unroll
        for (int j = 0; j < 16; ++j) sB[buf][(bnc + j) * 32 + bkr] = t.us[j];
    };

    loadTile(0, 0);
    __syncthreads();

    int buf = 0;
    for (int k0 = 0; k0 < K; k0 += 32) {
        if (k0 + 32 < K) loadTile(k0 + 32, buf ^ 1);   // fill other buffer

        const unsigned short* aB = &sA[buf][(wm * 32) * 32];
        const unsigned short* bB = &sB[buf][(wn * 64) * 32];
        v16bf a0 = load_frag_a(aB, lane);
        v16bf a1 = load_frag_a(aB + 16 * 32, lane);
        #pragma unroll
        for (int tn = 0; tn < 4; ++tn) {
            v16bf b = load_frag_b(bB + tn * 16 * 32, lane);
            acc[0][tn] = __builtin_amdgcn_wmma_f32_16x16x32_bf16(false, a0, false, b,
                                                                 (short)0, acc[0][tn],
                                                                 false, false);
            acc[1][tn] = __builtin_amdgcn_wmma_f32_16x16x32_bf16(false, a1, false, b,
                                                                 (short)0, acc[1][tn],
                                                                 false, false);
        }
        __syncthreads();
        buf ^= 1;
    }

    // ---- epilogue: C/D layout = VGPR r -> M=r (lanes 0-15) / M=8+r (lanes 16-31)
    const int m0 = bm + wm * 32;
    const int n0 = bn + wn * 64;
    const int nl = lane & 15;
    const int hi = (lane >> 4) * 8;
    #pragma unroll
    for (int tm = 0; tm < 2; ++tm) {
        #pragma unroll
        for (int tn = 0; tn < 4; ++tn) {
            const int col = n0 + tn * 16 + nl;
            #pragma unroll
            for (int r = 0; r < 8; ++r) {
                const int row = m0 + tm * 16 + hi + r;
                float v = acc[tm][tn][r];
                if (EPI >= 1) {
                    v += bias[col];
                    v = v > 0.f ? v : 0.f;
                }
                if (EPI == 2) {
                    if (row < M) Cf[(size_t)row * Nc + col] = v;
                } else {
                    Cb[(size_t)row * Nc + col] = f2bf(v);   // padded buffer: no guard
                }
            }
        }
    }
}

// ---------- driver ----------
extern "C" void kernel_launch(void* const* d_in, const int* in_sizes, int n_in,
                              void* d_out, int out_size, void* d_ws, size_t ws_size,
                              hipStream_t stream) {
    const float* x   = (const float*)d_in[0];
    const float* adj = (const float*)d_in[1];
    const float* W1  = (const float*)d_in[2];
    const float* b1  = (const float*)d_in[3];
    const float* W2  = (const float*)d_in[4];
    const float* b2  = (const float*)d_in[5];

    const int DOUT = in_sizes[3];              // b1 length = 512
    const int DIN  = in_sizes[2] / DOUT;       // W1 = DIN x DOUT
    const int N    = in_sizes[0] / DIN;        // x  = N x DIN
    const int P    = ((N + 127) / 128) * 128;  // padded node count (10112)

    // workspace carve-out (all offsets 256B aligned)
    auto up = [](size_t v) { return (v + 255) & ~(size_t)255; };
    char* w = (char*)d_ws;
    unsigned short* Mt      = (unsigned short*)w; w += up((size_t)P * P * 2);     // bf16 An^T (padded)
    unsigned short* Xb      = (unsigned short*)w; w += up((size_t)P * DIN * 2);
    unsigned short* W1b     = (unsigned short*)w; w += up((size_t)DIN * DOUT * 2);
    unsigned short* W2b     = (unsigned short*)w; w += up((size_t)DOUT * DOUT * 2);
    unsigned short* H1b     = (unsigned short*)w; w += up((size_t)P * DOUT * 2);
    unsigned short* Z1b     = (unsigned short*)w; w += up((size_t)P * DOUT * 2);
    unsigned short* H2b     = (unsigned short*)w; w += up((size_t)P * DOUT * 2);
    float*          dinv    = (float*)w;          w += up((size_t)N * 4);
    const int NPARTS = 32;
    float*          partial = (float*)w;          w += up((size_t)NPARTS * N * 4);
    float*          outF    = (float*)d_out;

    // 1) degree / dinv (deterministic two-stage reduction, no float atomics)
    {
        int rowsPer = (N + NPARTS - 1) / NPARTS;
        dim3 g((N + 255) / 256, NPARTS);
        colsum_partial<<<g, 256, 0, stream>>>(adj, partial, N, rowsPer);
        finish_dinv<<<(N + 255) / 256, 256, 0, stream>>>(partial, dinv, N, NPARTS);
    }
    // 2) normalized transposed adjacency -> bf16, zero-padded to P x P
    {
        int t = P / 32;
        norm_transpose<<<dim3(t, t), 256, 0, stream>>>(adj, dinv, Mt, N, P);
    }
    // 3) bf16 casts (row-zero-padded where used as GEMM A/B over padded K/M)
    cast_pad_bf16<<<2048, 256, 0, stream>>>(x,  Xb,  N,    DIN,  P);
    cast_pad_bf16<<<512,  256, 0, stream>>>(W1, W1b, DIN,  DOUT, DIN);
    cast_pad_bf16<<<512,  256, 0, stream>>>(W2, W2b, DOUT, DOUT, DOUT);

    dim3 grid(P / 128, DOUT / 128);

    // 4) H1 = x @ W1                 [P,DIN]x[DIN,DOUT] -> bf16 (pad rows produce 0)
    gemm_bf16_wmma<0><<<grid, 256, 0, stream>>>(Xb, W1b, nullptr, H1b, nullptr, N, DIN, DOUT);
    // 5) Z1 = relu(Mt @ H1 + b1)     [P,P]x[P,DOUT]     -> bf16 (big GEMM, K=P)
    gemm_bf16_wmma<1><<<grid, 256, 0, stream>>>(Mt, H1b, b1, Z1b, nullptr, N, P, DOUT);
    // 6) H2 = Z1 @ W2                                  -> bf16
    gemm_bf16_wmma<0><<<grid, 256, 0, stream>>>(Z1b, W2b, nullptr, H2b, nullptr, N, DOUT, DOUT);
    // 7) out = relu(Mt @ H2 + b2)    fp32 final output (row<M guarded)
    gemm_bf16_wmma<2><<<grid, 256, 0, stream>>>(Mt, H2b, b2, nullptr, outF, N, P, DOUT);
}
